// FragmentSimilarity_85452669321805
// MI455X (gfx1250) — matrix operand order repaired and checked by reading, not compiled, not run on hardware
//
#include <hip/hip_runtime.h>

// ---------------------------------------------------------------------------
// FragmentSimilarity on gfx1250 (MI455X): one workgroup per batch element.
// B=512, R=U=64, d=1024.  Memory-bound (AI ~16 FLOP/B), so: single pass over
// each input tensor, bf16 WMMA (v_wmma_f32_16x16x32_bf16) with f32 accum,
// h never materialized to global memory.
// ---------------------------------------------------------------------------

typedef __attribute__((ext_vector_type(16))) __bf16   v16bf;
typedef __attribute__((ext_vector_type(8)))  float    v8f;
typedef __attribute__((ext_vector_type(4)))  unsigned uint4v;

#define AST 72   // bf16 row stride for A-chunk / attn LDS tiles (144 B, 16B-aligned)
#define BST 72   // bf16 row stride for B-chunk LDS tile
#define SST 65   // f32 row stride for sim buffer

union FragA { unsigned u[8]; v16bf v; };
union FragB { uint4v  q[2]; v16bf v; };

__device__ __forceinline__ unsigned short f2bf(float x) {
    union { float f; unsigned u; } c; c.f = x;
    unsigned u = c.u;
    unsigned r = u + 0x7FFFu + ((u >> 16) & 1u);   // round-to-nearest-even
    return (unsigned short)(r >> 16);
}
__device__ __forceinline__ unsigned pack2(float a, float b) {
    return (unsigned)f2bf(a) | ((unsigned)f2bf(b) << 16);
}

__global__ __launch_bounds__(256)
void frag_sim_kernel(const float* __restrict__ rm,   // regions_m  [B,64,1024]
                     const float* __restrict__ wm,   // words_m    [B,64,1024]
                     const float* __restrict__ rr,   // regions_raw[B,64,1024]
                     const float* __restrict__ wr,   // words_raw  [B,64,1024]
                     float* __restrict__ out)
{
    __shared__ __align__(16) unsigned short aCh[64 * AST];   // bf16 chunk of regions_m
    __shared__ __align__(16) unsigned short bCh[64 * BST];   // bf16 chunk (words_m / words_raw^T)
    __shared__ __align__(16) unsigned short attnB[64 * AST]; // bf16 softmax weights
    __shared__ float simbuf[64 * SST];
    __shared__ float hhA[64], hvA[64], vvA[64], redbuf[64];

    const int b    = blockIdx.x;
    const int tid  = threadIdx.x;
    const int lane = tid & 31;
    const int wave = tid >> 5;
    const int lmod = lane & 15;
    const int lhi  = lane >> 4;            // 0: lanes 0-15, 1: lanes 16-31

    const size_t base = (size_t)b * 64 * 1024;

    if (tid < 64) { hhA[tid] = 0.f; hvA[tid] = 0.f; vvA[tid] = 0.f; }

    // Tile assignment: wave w owns output tiles 2w and 2w+1 of the 4x4 tile
    // grid -> both share the same tile-row ti, so one A fragment feeds 2 WMMAs.
    const int ti  = (wave * 2) >> 2;       // 0..3
    const int tj0 = (wave * 2) & 3;
    const int tj1 = tj0 + 1;
    const int rbase = ti * 16 + lhi * 8;   // first of this lane's 8 output rows

    // ======================= Phase 1: sim = Rm · Wm^T =======================
    v8f c0 = {0.f,0.f,0.f,0.f,0.f,0.f,0.f,0.f};
    v8f c1 = {0.f,0.f,0.f,0.f,0.f,0.f,0.f,0.f};

    for (int kc = 0; kc < 1024; kc += 64) {
        // cooperative load of 64x64 f32 chunks of both operands -> bf16 LDS
        #pragma unroll
        for (int i = 0; i < 4; ++i) {
            int idx = tid + i * 256;          // 0..1023 float4 slots
            int row = idx >> 4;
            int c4  = idx & 15;
            size_t g = base + (size_t)row * 1024 + kc + c4 * 4;
            float4 va = *(const float4*)&rm[g];
            float4 vb = *(const float4*)&wm[g];
            uint2 pa; pa.x = pack2(va.x, va.y); pa.y = pack2(va.z, va.w);
            uint2 pb; pb.x = pack2(vb.x, vb.y); pb.y = pack2(vb.z, vb.w);
            *(uint2*)&aCh[row * AST + c4 * 4] = pa;
            *(uint2*)&bCh[row * BST + c4 * 4] = pb;
        }
        __syncthreads();

        #pragma unroll
        for (int ks = 0; ks < 64; ks += 32) {
            FragA fa;                          // A: 16x32 bf16, 16 consecutive K/lane
            {
                const unsigned short* rowp = &aCh[(ti * 16 + lmod) * AST];
                int kb = ks + lhi * 16;
                #pragma unroll
                for (int p = 0; p < 8; ++p) {
                    int K = kb + 2 * p;
                    fa.u[p] = *(const unsigned*)&rowp[K];
                }
            }
            FragB fb0, fb1;                    // B: 32x16, 16 consecutive K/lane
            {
                int kb = ks + lhi * 16;
                const uint4v* p0 = (const uint4v*)&bCh[(tj0 * 16 + lmod) * BST + kb];
                fb0.q[0] = p0[0]; fb0.q[1] = p0[1];
                const uint4v* p1 = (const uint4v*)&bCh[(tj1 * 16 + lmod) * BST + kb];
                fb1.q[0] = p1[0]; fb1.q[1] = p1[1];
            }
            c0 = __builtin_amdgcn_wmma_f32_16x16x32_bf16(false, fa.v, false, fb0.v,
                                                         (short)0, c0, false, false);
            c1 = __builtin_amdgcn_wmma_f32_16x16x32_bf16(false, fa.v, false, fb1.v,
                                                         (short)0, c1, false, false);
        }
        __syncthreads();
    }

    // spill sim tiles (C/D layout: vgpr i -> row rbase+i, col = lane&15)
    #pragma unroll
    for (int i = 0; i < 8; ++i) {
        simbuf[(rbase + i) * SST + tj0 * 16 + lmod] = c0[i];
        simbuf[(rbase + i) * SST + tj1 * 16 + lmod] = c1[i];
    }
    __syncthreads();

    // ============ Phase 2: weak-match filter + softmax (lambda=9) ============
    if (tid < 64) {
        float* srow = &simbuf[tid * SST];
        float mean = 0.f;
        for (int u = 0; u < 64; ++u) mean += srow[u];
        mean *= (1.f / 64.f);
        float mx = -3.0e38f;
        for (int u = 0; u < 64; ++u) {
            float s = srow[u];
            float m = (s >= mean) ? s : -1e4f;
            mx = fmaxf(mx, m);
        }
        float sum = 0.f;
        for (int u = 0; u < 64; ++u) {
            float s = srow[u];
            float m = (s >= mean) ? s : -1e4f;
            float e = __expf((m - mx) * 9.0f);
            srow[u] = e;
            sum += e;
        }
        float inv = 1.f / sum;
        unsigned short* arow = &attnB[tid * AST];
        for (int u = 0; u < 64; ++u) arow[u] = f2bf(srow[u] * inv);
    }
    __syncthreads();

    // ====== Phase 3: h = attn · Wraw, fused cosine stats (no h in HBM) ======
    FragA a0, a1;                              // attn fragments, K=0..31 / 32..63
    {
        const unsigned short* rowp = &attnB[(ti * 16 + lmod) * AST];
        #pragma unroll
        for (int p = 0; p < 8; ++p) {
            int K = lhi * 16 + 2 * p;
            a0.u[p] = *(const unsigned*)&rowp[K];
            a1.u[p] = *(const unsigned*)&rowp[K + 32];
        }
    }
    float hh[8], hv[8], vv[8];
    #pragma unroll
    for (int i = 0; i < 8; ++i) { hh[i] = 0.f; hv[i] = 0.f; vv[i] = 0.f; }

    for (int nc = 0; nc < 1024; nc += 64) {
        __syncthreads();
        // words_raw chunk [64 u][64 n] -> bCh transposed (n-major) as bf16
        #pragma unroll
        for (int i = 0; i < 4; ++i) {
            int idx = tid + i * 256;
            int row = idx >> 4;                // u index
            int c4  = idx & 15;
            float4 vb = *(const float4*)&wr[base + (size_t)row * 1024 + nc + c4 * 4];
            bCh[(c4 * 4 + 0) * BST + row] = f2bf(vb.x);
            bCh[(c4 * 4 + 1) * BST + row] = f2bf(vb.y);
            bCh[(c4 * 4 + 2) * BST + row] = f2bf(vb.z);
            bCh[(c4 * 4 + 3) * BST + row] = f2bf(vb.w);
        }
        __syncthreads();

        v8f h0 = {0.f,0.f,0.f,0.f,0.f,0.f,0.f,0.f};
        v8f h1 = {0.f,0.f,0.f,0.f,0.f,0.f,0.f,0.f};
        #pragma unroll
        for (int ks = 0; ks < 64; ks += 32) {
            int kb = ks + lhi * 16;
            FragB fb0, fb1;
            const uint4v* p0 = (const uint4v*)&bCh[(tj0 * 16 + lmod) * BST + kb];
            fb0.q[0] = p0[0]; fb0.q[1] = p0[1];
            const uint4v* p1 = (const uint4v*)&bCh[(tj1 * 16 + lmod) * BST + kb];
            fb1.q[0] = p1[0]; fb1.q[1] = p1[1];
            const v16bf av = ks ? a1.v : a0.v;
            h0 = __builtin_amdgcn_wmma_f32_16x16x32_bf16(false, av, false, fb0.v,
                                                         (short)0, h0, false, false);
            h1 = __builtin_amdgcn_wmma_f32_16x16x32_bf16(false, av, false, fb1.v,
                                                         (short)0, h1, false, false);
        }

        // fold h tiles + regions_raw into per-lane cosine partials
        #pragma unroll
        for (int i = 0; i < 8; ++i) {
            size_t off = base + (size_t)(rbase + i) * 1024 + nc;
            float v0 = rr[off + tj0 * 16 + lmod];
            float v1 = rr[off + tj1 * 16 + lmod];
            float x0 = h0[i], x1 = h1[i];
            hh[i] += x0 * x0 + x1 * x1;
            hv[i] += x0 * v0 + x1 * v1;
            vv[i] += v0 * v0 + v1 * v1;
        }
    }

    #pragma unroll
    for (int i = 0; i < 8; ++i) {
        atomicAdd(&hhA[rbase + i], hh[i]);
        atomicAdd(&hvA[rbase + i], hv[i]);
        atomicAdd(&vvA[rbase + i], vv[i]);
    }
    __syncthreads();

    if (tid < 64) {
        float denom = (sqrtf(hhA[tid]) + 1e-8f) * (sqrtf(vvA[tid]) + 1e-8f);
        redbuf[tid] = hvA[tid] / denom;
    }
    __syncthreads();
    if (tid == 0) {
        float s = 0.f;
        for (int r = 0; r < 64; ++r) s += redbuf[r];
        out[b] = s * (1.f / 64.f);
    }
}

extern "C" void kernel_launch(void* const* d_in, const int* in_sizes, int n_in,
                              void* d_out, int out_size, void* d_ws, size_t ws_size,
                              hipStream_t stream) {
    const float* rm = (const float*)d_in[0];
    const float* wm = (const float*)d_in[1];
    const float* rr = (const float*)d_in[2];
    const float* wr = (const float*)d_in[3];
    float* out = (float*)d_out;
    const int B = in_sizes[0] / (64 * 1024);   // 512
    frag_sim_kernel<<<B, 256, 0, stream>>>(rm, wm, rr, wr, out);
}
